// ConsistencyLoss_30786325578234
// MI455X (gfx1250) — matrix operand order repaired and checked by reading, not compiled, not run on hardware
//
#include <hip/hip_runtime.h>
#include <math.h>

#define N_SLOTS 4096
#define DIM     8192
#define EPSN    1e-6f
#define BM      128
#define BN      128
#define KT      64
#define LDP     72                       // KT + 8 f16 pad -> 144B row stride (16B multiple)
#define NBLK    (N_SLOTS / BM)           // 32
#define NUPPER  (NBLK * (NBLK + 1) / 2)  // 528 upper-triangular blocks

typedef __attribute__((ext_vector_type(16))) _Float16 v16h;
typedef __attribute__((ext_vector_type(8)))  _Float16 v8h;
typedef __attribute__((ext_vector_type(8)))  float    v8f;

union H16 { v16h v; v8h h[2]; };

union __align__(16) SharedU {
    struct { _Float16 A[BM][LDP]; _Float16 B[BN][LDP]; } g;  // 36864 B (GEMM tiles)
    struct { float sI[128][32]; float sJ[128][8]; } r;       // 20480 B (row/col partials)
    float red[256];
};

// ---------------------------------------------------------------------------
// Inverse per-row norms from f16-rounded data (consistent with the GEMM).
// Storing 1/max(||s||, eps) turns all epilogue divides into multiplies.
// ---------------------------------------------------------------------------
__global__ __launch_bounds__(256) void norms_kernel(const float* __restrict__ slots,
                                                    float* __restrict__ invnorm) {
    __shared__ float red[256];
    const int row = blockIdx.x;
    const float* p = slots + (size_t)row * DIM;
    float s = 0.f;
    for (int k = threadIdx.x; k < DIM; k += 256) {
        float x = (float)(_Float16)p[k];   // round through f16 like the GEMM
        s += x * x;
    }
    red[threadIdx.x] = s;
    __syncthreads();
    for (int off = 128; off > 0; off >>= 1) {
        if (threadIdx.x < off) red[threadIdx.x] += red[threadIdx.x + off];
        __syncthreads();
    }
    if (threadIdx.x == 0) invnorm[row] = 1.0f / fmaxf(sqrtf(red[0]), EPSN);
}

// ---------------------------------------------------------------------------
// One-shot fp32 -> f16 conversion of the whole matrix (used when ws allows).
// ---------------------------------------------------------------------------
__global__ __launch_bounds__(256) void convert_kernel(const float* __restrict__ src,
                                                      _Float16* __restrict__ dst) {
    const size_t idx = ((size_t)blockIdx.x * blockDim.x + threadIdx.x) * 8;
    float4 a = *reinterpret_cast<const float4*>(src + idx);
    float4 b = *reinterpret_cast<const float4*>(src + idx + 4);
    v8h p;
    p[0]=(_Float16)a.x; p[1]=(_Float16)a.y; p[2]=(_Float16)a.z; p[3]=(_Float16)a.w;
    p[4]=(_Float16)b.x; p[5]=(_Float16)b.y; p[6]=(_Float16)b.z; p[7]=(_Float16)b.w;
    *reinterpret_cast<v8h*>(dst + idx) = p;
}

// Stage 32 consecutive K elements of one tile row into LDS.
__device__ __forceinline__ void stage32(const float* __restrict__ g, _Float16* __restrict__ d) {
    const float4* ga = reinterpret_cast<const float4*>(g);
#pragma unroll
    for (int c = 0; c < 4; ++c) {
        float4 x = ga[2*c], y = ga[2*c + 1];
        v8h p;
        p[0]=(_Float16)x.x; p[1]=(_Float16)x.y; p[2]=(_Float16)x.z; p[3]=(_Float16)x.w;
        p[4]=(_Float16)y.x; p[5]=(_Float16)y.y; p[6]=(_Float16)y.z; p[7]=(_Float16)y.w;
        *(v8h*)(d + 8*c) = p;
    }
}
__device__ __forceinline__ void stage32(const _Float16* __restrict__ g, _Float16* __restrict__ d) {
    const float4* ga = reinterpret_cast<const float4*>(g);   // 8 f16 per float4
#pragma unroll
    for (int c = 0; c < 2; ++c)
        *reinterpret_cast<float4*>(d + 16*c) = ga[c];
}

// ---------------------------------------------------------------------------
// Tiled Gram body: 128x128 output tile via 8 waves, each wave owning 2x4
// v_wmma_f32_16x16x32_f16 accumulators; KT=64 (2 WMMA k-steps per barrier).
// ---------------------------------------------------------------------------
template <typename SRC>
__device__ __forceinline__ void gemm_tile(const SRC* __restrict__ mat,
                                          int bi, int bj,
                                          _Float16 (*As)[LDP], _Float16 (*Bs)[LDP],
                                          v8f acc[2][4]) {
    const int tid  = threadIdx.x;
    const int lane = tid & 31;
    const int wave = tid >> 5;
    const int wr   = wave >> 1;   // 0..3 : wave row
    const int wc   = wave & 1;    // 0..1 : wave col
    const int ln   = lane & 15;
    const int lh   = lane >> 4;   // lane half

    const v8f zero = {0.f,0.f,0.f,0.f,0.f,0.f,0.f,0.f};
#pragma unroll
    for (int tm = 0; tm < 2; ++tm)
#pragma unroll
        for (int tn = 0; tn < 4; ++tn) acc[tm][tn] = zero;

    const int lrow  = tid >> 1;   // 0..127 : tile row this thread stages
    const int lhalf = tid & 1;    // 0..1   : which 32-element K-half
    const size_t arow = (size_t)(bi * BM + lrow) * DIM + lhalf * 32;
    const size_t brow = (size_t)(bj * BN + lrow) * DIM + lhalf * 32;

    for (int kt = 0; kt < DIM; kt += KT) {
        stage32(mat + arow + kt, &As[lrow][lhalf * 32]);
        stage32(mat + brow + kt, &Bs[lrow][lhalf * 32]);
        if (kt + KT < DIM) {
            __builtin_prefetch((const void*)(mat + arow + kt + KT), 0, 1);
            __builtin_prefetch((const void*)(mat + brow + kt + KT), 0, 1);
        }
        __syncthreads();

#pragma unroll
        for (int ks = 0; ks < KT; ks += 32) {
            // ISA 16-bit A layout: lanes 0-15 hold K 0-7/16-23, lanes 16-31 hold K 8-15/24-31
            H16 af[2], bf[4];
#pragma unroll
            for (int tm = 0; tm < 2; ++tm) {
                const int r = wr*32 + tm*16 + ln;
                af[tm].h[0] = *(const v8h*)&As[r][ks + 8*lh];
                af[tm].h[1] = *(const v8h*)&As[r][ks + 16 + 8*lh];
            }
            // B (32x16) layout: lanes 0-15 hold K 0-7/16-23, lanes 16-31 hold K 8-15/24-31
#pragma unroll
            for (int tn = 0; tn < 4; ++tn) {
                const int c = wc*64 + tn*16 + ln;
                bf[tn].h[0] = *(const v8h*)&Bs[c][ks + 8*lh];
                bf[tn].h[1] = *(const v8h*)&Bs[c][ks + 16 + 8*lh];
            }
#pragma unroll
            for (int tm = 0; tm < 2; ++tm)
#pragma unroll
                for (int tn = 0; tn < 4; ++tn)
                    acc[tm][tn] = __builtin_amdgcn_wmma_f32_16x16x32_f16(
                        false, af[tm].v, false, bf[tn].v, (short)0, acc[tm][tn], false, false);
        }
        __syncthreads();
    }
}

__device__ __forceinline__ void block_to_tiles(int* bi, int* bj) {
    int b = blockIdx.x, r = 0;
    while (b >= NBLK - r) { b -= NBLK - r; ++r; }
    *bi = r; *bj = r + b;
}

// ---------------------------------------------------------------------------
// Pass 1: E = exp(S/T) on upper-triangular tiles; deterministic per-(row,
// col-tile) partials. Diagonal blocks emit full row sums (incl. k==i);
// off-diagonal blocks also credit the symmetric side (E is symmetric).
// ---------------------------------------------------------------------------
template <typename SRC>
__global__ __launch_bounds__(256) void pass1_kernel(const SRC* __restrict__ mat,
                                                    const float* __restrict__ invnorm,
                                                    const float* __restrict__ temperature,
                                                    float* __restrict__ rowpart) {
    __shared__ SharedU sh;
    int bi, bj; block_to_tiles(&bi, &bj);

    v8f acc[2][4];
    gemm_tile(mat, bi, bj, sh.g.A, sh.g.B, acc);

    const int tid = threadIdx.x;
    const int lane = tid & 31, wave = tid >> 5;
    const int wr = wave >> 1, wc = wave & 1;
    const int ln = lane & 15, lh = lane >> 4;
    const float invT = 1.0f / temperature[0];

    const int ibase = bi*BM + wr*32 + 8*lh;   // + tm*16 + v
    const int jbase = bj*BN + wc*64 + ln;     // + tn*16

    float inj[4];
#pragma unroll
    for (int tn = 0; tn < 4; ++tn) inj[tn] = invnorm[jbase + tn*16] * invT;

#pragma unroll
    for (int tm = 0; tm < 2; ++tm)
#pragma unroll
        for (int v = 0; v < 8; ++v) {
            const float ini = invnorm[ibase + tm*16 + v];
#pragma unroll
            for (int tn = 0; tn < 4; ++tn)
                acc[tm][tn][v] = __expf(acc[tm][tn][v] * ini * inj[tn]);  // stash E
        }

    // Stage per-lane partials into unique LDS slots (deterministic).
#pragma unroll
    for (int tm = 0; tm < 2; ++tm)
#pragma unroll
        for (int v = 0; v < 8; ++v) {
            float pI = 0.f;
#pragma unroll
            for (int tn = 0; tn < 4; ++tn) pI += acc[tm][tn][v];
            sh.r.sI[wr*32 + tm*16 + 8*lh + v][wc*16 + ln] = pI;
        }
#pragma unroll
    for (int tn = 0; tn < 4; ++tn) {
        float pJ = 0.f;
#pragma unroll
        for (int tm = 0; tm < 2; ++tm)
#pragma unroll
            for (int v = 0; v < 8; ++v) pJ += acc[tm][tn][v];
        sh.r.sJ[wc*64 + tn*16 + ln][wr*2 + lh] = pJ;
    }
    __syncthreads();

    if (tid < 128) {
        float s = 0.f;
#pragma unroll
        for (int c = 0; c < 32; ++c) s += sh.r.sI[tid][c];
        rowpart[(size_t)(bi*BM + tid) * NBLK + bj] = s;
    } else if (bi != bj) {
        const int c = tid - 128;
        float s = 0.f;
#pragma unroll
        for (int k = 0; k < 8; ++k) s += sh.r.sJ[c][k];
        rowpart[(size_t)(bj*BN + c) * NBLK + bi] = s;
    }
}

__global__ __launch_bounds__(256) void rowsum_kernel(const float* __restrict__ rowpart,
                                                     float* __restrict__ rowsum) {
    const int i = blockIdx.x * blockDim.x + threadIdx.x;
    if (i < N_SLOTS) {
        const float* p = rowpart + (size_t)i * NBLK;
        float s = 0.f;
#pragma unroll
        for (int c = 0; c < NBLK; ++c) s += p[c];
        rowsum[i] = s;
    }
}

// ---------------------------------------------------------------------------
// Pass 2: recompute identical tiles (bitwise-identical E) and accumulate
// (log(rowsum[i] - E) - logit) * (j - i) over i < j with fixed-order reduce.
// ---------------------------------------------------------------------------
template <typename SRC>
__global__ __launch_bounds__(256) void pass2_kernel(const SRC* __restrict__ mat,
                                                    const float* __restrict__ invnorm,
                                                    const float* __restrict__ temperature,
                                                    const float* __restrict__ rowsum,
                                                    float* __restrict__ lossPart) {
    __shared__ SharedU sh;
    int bi, bj; block_to_tiles(&bi, &bj);

    v8f acc[2][4];
    gemm_tile(mat, bi, bj, sh.g.A, sh.g.B, acc);

    const int tid = threadIdx.x;
    const int lane = tid & 31, wave = tid >> 5;
    const int wr = wave >> 1, wc = wave & 1;
    const int ln = lane & 15, lh = lane >> 4;
    const float invT = 1.0f / temperature[0];

    const int ibase = bi*BM + wr*32 + 8*lh;
    const int jbase = bj*BN + wc*64 + ln;

    float inj[4];
#pragma unroll
    for (int tn = 0; tn < 4; ++tn) inj[tn] = invnorm[jbase + tn*16] * invT;

    float lsum = 0.f;
#pragma unroll
    for (int tm = 0; tm < 2; ++tm)
#pragma unroll
        for (int v = 0; v < 8; ++v) {
            const int i = ibase + tm*16 + v;
            const float ini = invnorm[i];
            const float rsi = rowsum[i];
#pragma unroll
            for (int tn = 0; tn < 4; ++tn) {
                const int j = jbase + tn*16;
                if (i < j) {
                    float logit = acc[tm][tn][v] * ini * inj[tn];
                    float e = __expf(logit);
                    float den = rsi - e;
                    lsum += (__logf(den) - logit) * (float)(j - i);
                }
            }
        }

    __syncthreads();
    sh.red[tid] = lsum;
    __syncthreads();
    for (int off = 128; off > 0; off >>= 1) {
        if (tid < off) sh.red[tid] += sh.red[tid + off];
        __syncthreads();
    }
    if (tid == 0) lossPart[blockIdx.x] = sh.red[0];
}

__global__ void finalize_kernel(const float* __restrict__ lossPart,
                                const int* __restrict__ length,
                                float* __restrict__ out) {
    if (threadIdx.x == 0 && blockIdx.x == 0) {
        double t = 0.0;
        for (int k = 0; k < NUPPER; ++k) t += (double)lossPart[k];
        const int len = length[0];
        out[0] = (float)t;
        out[1] = (float)(t / ((double)(len - 1) * (double)(len - 1) * 0.5));
    }
}

extern "C" void kernel_launch(void* const* d_in, const int* in_sizes, int n_in,
                              void* d_out, int out_size, void* d_ws, size_t ws_size,
                              hipStream_t stream) {
    (void)in_sizes; (void)n_in; (void)out_size;
    const float* slots       = (const float*)d_in[0];
    const int*   length      = (const int*)d_in[1];
    const float* temperature = (const float*)d_in[2];
    float* out = (float*)d_out;

    // Workspace: invnorm[N] | rowsum[N] | rowpart[N][NBLK] | lossPart | (pad) | f16 slots copy
    float* invnorm  = (float*)d_ws;
    float* rowsum   = invnorm + N_SLOTS;
    float* rowpart  = rowsum + N_SLOTS;
    float* lossPart = rowpart + (size_t)N_SLOTS * NBLK;
    size_t smallBytes = ((size_t)(N_SLOTS + N_SLOTS + N_SLOTS * NBLK + NUPPER) * sizeof(float)
                         + 4095) & ~(size_t)4095;
    _Float16* h16 = (_Float16*)((char*)d_ws + smallBytes);
    const size_t f16Bytes = (size_t)N_SLOTS * DIM * sizeof(_Float16);
    const bool useF16 = ws_size >= smallBytes + f16Bytes;   // ws_size is fixed -> deterministic

    norms_kernel<<<N_SLOTS, 256, 0, stream>>>(slots, invnorm);

    if (useF16) {
        convert_kernel<<<(int)(((size_t)N_SLOTS * DIM / 8) / 256), 256, 0, stream>>>(slots, h16);
        pass1_kernel<_Float16><<<NUPPER, 256, 0, stream>>>(h16, invnorm, temperature, rowpart);
        rowsum_kernel<<<(N_SLOTS + 255) / 256, 256, 0, stream>>>(rowpart, rowsum);
        pass2_kernel<_Float16><<<NUPPER, 256, 0, stream>>>(h16, invnorm, temperature, rowsum, lossPart);
    } else {
        pass1_kernel<float><<<NUPPER, 256, 0, stream>>>(slots, invnorm, temperature, rowpart);
        rowsum_kernel<<<(N_SLOTS + 255) / 256, 256, 0, stream>>>(rowpart, rowsum);
        pass2_kernel<float><<<NUPPER, 256, 0, stream>>>(slots, invnorm, temperature, rowsum, lossPart);
    }
    finalize_kernel<<<1, 32, 0, stream>>>(lossPart, length, out);
}
